// MultiHeadAttention_80719615361625
// MI455X (gfx1250) — compile-verified, matrix operand order checked by reading, not run on hardware
//
#include <hip/hip_runtime.h>
#include <hip/hip_bf16.h>

// ---------------------------------------------------------------------------
// MultiHeadAttention forward for MI455X (gfx1250, wave32, WMMA bf16 path).
// B=2, S=2048, E=1024, H=16, D=64.
//
// Pipeline:
//   1) proj_gemm<fp32 in, bf16 out> x3 : Q/K/V = x @ W^T + b      (ws)
//   2) scores  : attn_raw = Qh @ Kh^T / 8  (async-LDS K tile, WMMA, fp32 out)
//   3) softmax : in-place row softmax over last dim (2048)
//   4) attnv   : Oh = attn @ Vh            (prefetched fp32 A, bf16 ws out)
//   5) proj_gemm<bf16 in, fp32 out> : out = O @ Wo^T + bo (async-LDS A tile)
//
// ws layout (bf16): Qb | Kb | Vb | Ob, each 4096*1024 elements = 32 MB total.
// ---------------------------------------------------------------------------

typedef __attribute__((ext_vector_type(16))) __bf16 v16bf;
typedef __attribute__((ext_vector_type(8)))  __bf16 v8bf;
typedef __attribute__((ext_vector_type(8)))  float  v8f;

#define E_DIM 1024
#define NHEADS 16
#define HDIM 64
#define BATCH 2
#define SEQ 2048
#define MTOT (BATCH * SEQ)   // 4096

union ABFrag {
  v16bf v;
  v8bf  h[2];
  __bf16 e[16];
};

// 16x32 bf16 A/B fragment from a row-major tile. CDNA5 16-bit layout:
// lane m in [0,16) holds K {half*8..+7} and {16+half*8..+7}; B mirrors, lane=N.
__device__ __forceinline__ v16bf ld_frag(const __bf16* rowptr, int half) {
  ABFrag u;
  u.h[0] = *(const v8bf*)(rowptr + half * 8);
  u.h[1] = *(const v8bf*)(rowptr + half * 8 + 16);
  return u.v;
}

__device__ __forceinline__ v8f wmma_bf16(v16bf a, v16bf b, v8f c) {
  return __builtin_amdgcn_wmma_f32_16x16x32_bf16(false, a, false, b,
                                                 (short)0, c, false, false);
}

// Async 16-byte global->LDS copy (gfx1250 GLOBAL_LOAD_ASYNC_TO_LDS_B128,
// tracked with ASYNCcnt). The LDS address is ptrtoint of the real shared-mem
// pointer: low 32 bits of a generic pointer to LDS are the LDS byte address,
// and the capture forces LLVM to assume the asm writes the object.
__device__ __forceinline__ void async_copy_b128(const void* gptr, void* lptr) {
  unsigned lds_addr = (unsigned)(size_t)lptr;
  asm volatile("global_load_async_to_lds_b128 %0, %1, off"
               :: "v"(lds_addr), "v"(gptr)
               : "memory");
}
__device__ __forceinline__ void async_wait0() {
  asm volatile("s_wait_asynccnt 0x0" ::: "memory");
}

// ---------------------------------------------------------------------------
// Projection GEMM: Y[4096,1024] = X[4096,1024] @ W[1024,1024]^T + bias.
// Block tile 128(M) x 64(N), K-step 32, 8 waves; wave w owns rows
// [m0+16w, m0+16w+16) and 4 WMMA tiles along N.
// ---------------------------------------------------------------------------
template <bool A_BF16, bool OUT_F32>
__global__ __launch_bounds__(256)
void proj_gemm_kernel(const void* __restrict__ Xin,
                      const float* __restrict__ W,
                      const float* __restrict__ bias,
                      void* __restrict__ Yout) {
  constexpr int K = E_DIM;
  constexpr int N = E_DIM;
  __shared__ __bf16 As[128][32];   // 8 KB
  __shared__ __bf16 Bs[64][32];    // 4 KB

  const int tid  = threadIdx.x;
  const int wv   = tid >> 5;
  const int lane = tid & 31;
  const int half = lane >> 4;
  const int r    = lane & 15;
  const int m0   = blockIdx.y * 128;
  const int n0   = blockIdx.x * 64;

  v8f acc[4] = {};

  for (int k0 = 0; k0 < K; k0 += 32) {
    __syncthreads();
    // --- stage A tile (128 x 32) ---
    if constexpr (A_BF16) {
      // Pure bf16 copy: use CDNA5 async global->LDS transfers.
      const __bf16* Xb = (const __bf16*)Xin;
      #pragma unroll
      for (int i = 0; i < 2; ++i) {
        int c = tid + i * 256;                    // 512 v8bf chunks
        int row = c >> 2, s = (c & 3) * 8;
        async_copy_b128(Xb + (size_t)(m0 + row) * K + k0 + s, &As[row][s]);
      }
    } else {
      const float* Xf = (const float*)Xin;
      #pragma unroll
      for (int i = 0; i < 4; ++i) {
        int vv = tid + i * 256;                   // 1024 float4 chunks
        int row = vv >> 3, c4 = (vv & 7) * 4;
        float4 f = *(const float4*)(Xf + (size_t)(m0 + row) * K + k0 + c4);
        As[row][c4 + 0] = (__bf16)f.x;
        As[row][c4 + 1] = (__bf16)f.y;
        As[row][c4 + 2] = (__bf16)f.z;
        As[row][c4 + 3] = (__bf16)f.w;
      }
    }
    // --- stage B tile (64 x 32); W rows are contiguous in K ---
    #pragma unroll
    for (int i = 0; i < 2; ++i) {
      int vv = tid + i * 256;                     // 512 float4 chunks
      int row = vv >> 3, c4 = (vv & 7) * 4;
      float4 f = *(const float4*)(W + (size_t)(n0 + row) * K + k0 + c4);
      Bs[row][c4 + 0] = (__bf16)f.x;
      Bs[row][c4 + 1] = (__bf16)f.y;
      Bs[row][c4 + 2] = (__bf16)f.z;
      Bs[row][c4 + 3] = (__bf16)f.w;
    }
    if constexpr (A_BF16) async_wait0();
    __syncthreads();

    // Load all fragments first, then a back-to-back WMMA burst.
    v16bf a = ld_frag(&As[wv * 16 + r][0], half);
    v16bf bfr[4];
    #pragma unroll
    for (int nt = 0; nt < 4; ++nt) bfr[nt] = ld_frag(&Bs[nt * 16 + r][0], half);
    #pragma unroll
    for (int nt = 0; nt < 4; ++nt) acc[nt] = wmma_bf16(a, bfr[nt], acc[nt]);
  }

  // C layout: VGPR r2, lanes 0-15 -> M=r2, lanes 16-31 -> M=8+r2; N=lane&15.
  #pragma unroll
  for (int nt = 0; nt < 4; ++nt) {
    int col = n0 + nt * 16 + r;
    float bv = bias[col];
    int rowbase = m0 + wv * 16 + half * 8;
    #pragma unroll
    for (int r2 = 0; r2 < 8; ++r2) {
      float val = acc[nt][r2] + bv;
      size_t idx = (size_t)(rowbase + r2) * N + col;
      if constexpr (OUT_F32) ((float*)Yout)[idx]  = val;
      else                   ((__bf16*)Yout)[idx] = (__bf16)val;
    }
  }
}

// ---------------------------------------------------------------------------
// scores[z,q,k] = (Qh[q,:] . Kh[k,:]) / 8, z = b*H+h.
// Block tile 128(q) x 128(k); D=64 -> 2 K-steps. The K tile (16 KB, pure
// bf16 copy) is staged with async global->LDS b128 transfers (ASYNCcnt),
// fenced by s_wait_asynccnt before the workgroup barrier.
// ---------------------------------------------------------------------------
__global__ __launch_bounds__(256)
void scores_kernel(const __bf16* __restrict__ Qb, const __bf16* __restrict__ Kb,
                   float* __restrict__ attn) {
  __shared__ __bf16 Ks[128][64];   // 16 KB

  const int tid  = threadIdx.x;
  const int wv   = tid >> 5;
  const int lane = tid & 31;
  const int half = lane >> 4;
  const int r    = lane & 15;
  const int n0   = blockIdx.x * 128;
  const int m0   = blockIdx.y * 128;
  const int z    = blockIdx.z;               // b*H + h
  const int b    = z >> 4, h = z & 15;
  const size_t rowbase = (size_t)b * SEQ;
  const size_t hcol    = (size_t)h * HDIM;

  // Async bulk copy: 1024 chunks of 16 B, 4 per thread.
  const __bf16* ksrc = Kb + (rowbase + n0) * E_DIM + hcol;
  #pragma unroll
  for (int i = 0; i < 4; ++i) {
    int c = tid + i * 256;
    int row = c >> 3, s = (c & 7) * 8;
    async_copy_b128(ksrc + (size_t)row * E_DIM + s, &Ks[row][s]);
  }
  async_wait0();
  __syncthreads();

  v8f acc[8] = {};
  const __bf16* qrow = Qb + (rowbase + m0 + wv * 16 + r) * E_DIM + hcol;
  #pragma unroll
  for (int ds = 0; ds < 2; ++ds) {
    ABFrag ua;
    ua.h[0] = *(const v8bf*)(qrow + ds * 32 + half * 8);
    ua.h[1] = *(const v8bf*)(qrow + ds * 32 + half * 8 + 16);
    v16bf bfr[8];
    #pragma unroll
    for (int nt = 0; nt < 8; ++nt)
      bfr[nt] = ld_frag(&Ks[nt * 16 + r][ds * 32], half);
    #pragma unroll
    for (int nt = 0; nt < 8; ++nt) acc[nt] = wmma_bf16(ua.v, bfr[nt], acc[nt]);
  }

  const float scale = 0.125f;                // 1/sqrt(64)
  #pragma unroll
  for (int nt = 0; nt < 8; ++nt) {
    int col = n0 + nt * 16 + r;
    int qb  = m0 + wv * 16 + half * 8;
    #pragma unroll
    for (int r2 = 0; r2 < 8; ++r2)
      attn[((size_t)z * SEQ + qb + r2) * SEQ + col] = acc[nt][r2] * scale;
  }
}

// ---------------------------------------------------------------------------
// In-place row softmax over 2048 elements; one 256-thread block per row.
// ---------------------------------------------------------------------------
__global__ __launch_bounds__(256)
void softmax_kernel(float* __restrict__ attn) {
  __shared__ float red[256];
  const int t = threadIdx.x;
  float* p = attn + (size_t)blockIdx.x * SEQ;

  float4 x0 = *(float4*)(p + t * 8);
  float4 x1 = *(float4*)(p + t * 8 + 4);
  float m = fmaxf(fmaxf(fmaxf(x0.x, x0.y), fmaxf(x0.z, x0.w)),
                  fmaxf(fmaxf(x1.x, x1.y), fmaxf(x1.z, x1.w)));
  red[t] = m;
  __syncthreads();
  for (int s = 128; s > 0; s >>= 1) {
    if (t < s) red[t] = fmaxf(red[t], red[t + s]);
    __syncthreads();
  }
  const float rm = red[0];
  __syncthreads();

  x0.x = __expf(x0.x - rm); x0.y = __expf(x0.y - rm);
  x0.z = __expf(x0.z - rm); x0.w = __expf(x0.w - rm);
  x1.x = __expf(x1.x - rm); x1.y = __expf(x1.y - rm);
  x1.z = __expf(x1.z - rm); x1.w = __expf(x1.w - rm);
  red[t] = x0.x + x0.y + x0.z + x0.w + x1.x + x1.y + x1.z + x1.w;
  __syncthreads();
  for (int s = 128; s > 0; s >>= 1) {
    if (t < s) red[t] += red[t + s];
    __syncthreads();
  }
  const float inv = 1.0f / red[0];

  x0.x *= inv; x0.y *= inv; x0.z *= inv; x0.w *= inv;
  x1.x *= inv; x1.y *= inv; x1.z *= inv; x1.w *= inv;
  *(float4*)(p + t * 8)     = x0;
  *(float4*)(p + t * 8 + 4) = x1;
}

// ---------------------------------------------------------------------------
// Oh[z, q, d] = sum_k attn[z,q,k] * Vh[z,k,d].  M=2048, N=64, Kdim=2048.
// Block tile 128(q) x 64(d); V tile transposed into LDS as [d][k].
// attn rows (the 512 MB stream) get a one-tile-ahead global prefetch.
// ---------------------------------------------------------------------------
__global__ __launch_bounds__(256)
void attnv_kernel(const float* __restrict__ attn, const __bf16* __restrict__ Vb,
                  __bf16* __restrict__ Ob) {
  __shared__ __bf16 Vs[64][32];   // [d][k], 4 KB

  const int tid  = threadIdx.x;
  const int wv   = tid >> 5;
  const int lane = tid & 31;
  const int half = lane >> 4;
  const int r    = lane & 15;
  const int m0   = blockIdx.x * 128;
  const int z    = blockIdx.y;               // b*H + h
  const int b    = z >> 4, h = z & 15;
  const size_t rowbase = (size_t)b * SEQ;
  const size_t hcol    = (size_t)h * HDIM;

  v8f acc[4] = {};
  const float* arow = attn + ((size_t)z * SEQ + m0 + wv * 16 + r) * SEQ;

  for (int k0 = 0; k0 < SEQ; k0 += 32) {
    __syncthreads();
    {   // stage V[k0..k0+31, :64] transposed: 32 k-rows x 8 d-chunks
      int k = tid >> 3, s = (tid & 7) * 8;
      v8bf v = *(const v8bf*)(Vb + (rowbase + k0 + k) * E_DIM + hcol + s);
      #pragma unroll
      for (int j = 0; j < 8; ++j) Vs[s + j][k] = v[j];
    }
    __syncthreads();

    if (k0 + 32 < SEQ)  // pull next attn chunk toward the caches early
      __builtin_prefetch(arow + k0 + 32 + half * 8, 0, 3);

    // A fragment: 16 attn probs (fp32) converted to bf16 in layout order.
    ABFrag ua;
    #pragma unroll
    for (int g = 0; g < 2; ++g) {
      const float* ap = arow + k0 + g * 16 + half * 8;
      float4 f0 = *(const float4*)(ap);
      float4 f1 = *(const float4*)(ap + 4);
      int o = g * 8;
      ua.e[o + 0] = (__bf16)f0.x; ua.e[o + 1] = (__bf16)f0.y;
      ua.e[o + 2] = (__bf16)f0.z; ua.e[o + 3] = (__bf16)f0.w;
      ua.e[o + 4] = (__bf16)f1.x; ua.e[o + 5] = (__bf16)f1.y;
      ua.e[o + 6] = (__bf16)f1.z; ua.e[o + 7] = (__bf16)f1.w;
    }
    v16bf bfr[4];
    #pragma unroll
    for (int nt = 0; nt < 4; ++nt) bfr[nt] = ld_frag(&Vs[nt * 16 + r][0], half);
    #pragma unroll
    for (int nt = 0; nt < 4; ++nt) acc[nt] = wmma_bf16(ua.v, bfr[nt], acc[nt]);
  }

  #pragma unroll
  for (int nt = 0; nt < 4; ++nt) {
    int d  = nt * 16 + r;
    int qb = m0 + wv * 16 + half * 8;
    #pragma unroll
    for (int r2 = 0; r2 < 8; ++r2)
      Ob[(rowbase + qb + r2) * E_DIM + hcol + d] = (__bf16)acc[nt][r2];
  }
}

// ---------------------------------------------------------------------------
extern "C" void kernel_launch(void* const* d_in, const int* in_sizes, int n_in,
                              void* d_out, int out_size, void* d_ws, size_t ws_size,
                              hipStream_t stream) {
  const float* q  = (const float*)d_in[0];
  const float* k  = (const float*)d_in[1];
  const float* v  = (const float*)d_in[2];
  const float* Wq = (const float*)d_in[3];
  const float* bq = (const float*)d_in[4];
  const float* Wk = (const float*)d_in[5];
  const float* bk = (const float*)d_in[6];
  const float* Wv = (const float*)d_in[7];
  const float* bv = (const float*)d_in[8];
  const float* Wo = (const float*)d_in[9];
  const float* bo = (const float*)d_in[10];

  float* out  = (float*)d_out;                      // [B,S,E]
  float* attn = out + (size_t)MTOT * E_DIM;         // [B,H,S,S]

  __bf16* Qb = (__bf16*)d_ws;                       // 8 MB each
  __bf16* Kb = Qb + (size_t)MTOT * E_DIM;
  __bf16* Vf = Kb + (size_t)MTOT * E_DIM;
  __bf16* Ob = Vf + (size_t)MTOT * E_DIM;

  dim3 gp(E_DIM / 64, MTOT / 128);                  // (16, 32)
  proj_gemm_kernel<false, false><<<gp, 256, 0, stream>>>(q, Wq, bq, Qb);
  proj_gemm_kernel<false, false><<<gp, 256, 0, stream>>>(k, Wk, bk, Kb);
  proj_gemm_kernel<false, false><<<gp, 256, 0, stream>>>(v, Wv, bv, Vf);

  scores_kernel<<<dim3(SEQ / 128, SEQ / 128, BATCH * NHEADS), 256, 0, stream>>>(
      Qb, Kb, attn);
  softmax_kernel<<<dim3(BATCH * NHEADS * SEQ), 256, 0, stream>>>(attn);
  attnv_kernel<<<dim3(SEQ / 128, BATCH * NHEADS), 256, 0, stream>>>(attn, Vf, Ob);

  proj_gemm_kernel<true, true><<<gp, 256, 0, stream>>>(Ob, Wo, bo, out);
}
